// _LightVLACore_17214228922631
// MI455X (gfx1250) — compile-verified, matrix operand order checked by reading, not run on hardware
//
#include <hip/hip_runtime.h>

// ---------------------------------------------------------------------------
// LightVLA core, CDNA5 (gfx1250) implementation.
// B=16, N=768, T=48, D=2048. Output = one_hot(argmax_m(q_raw . patches_n)),
// since hard + soft - stop_grad(soft) == hard numerically, and argmax is
// invariant to the positive rms-norm row scale and the 1/sqrt(D) factor.
// All GEMMs run on v_wmma_f32_16x16x32_bf16 (wave32).
// Score kernel uses 2x4 register blocking (32 rows x 64 cols per wave):
// 8 WMMAs per 6 fragment loads -> compute-bound on the matrix pipes.
// ---------------------------------------------------------------------------

#define BB 16
#define NN 768
#define TT 48
#define DD 2048

typedef __attribute__((ext_vector_type(16))) __bf16 v16bf;
typedef __attribute__((ext_vector_type(4)))  __bf16 v4bf;
typedef __attribute__((ext_vector_type(8)))  float  v8f;

__device__ __forceinline__ __bf16 f2bf(float f) {
  union { float f; unsigned int u; } v; v.f = f;
  unsigned int r = v.u + 0x7FFFu + ((v.u >> 16) & 1u);   // round-nearest-even
  unsigned short h = (unsigned short)(r >> 16);
  __bf16 o; __builtin_memcpy(&o, &h, 2); return o;
}

__device__ __forceinline__ v8f wmma_bf16(v16bf a, v16bf b, v8f c) {
  return __builtin_amdgcn_wmma_f32_16x16x32_bf16(
      /*neg_a=*/false, a, /*neg_b=*/false, b,
      /*c_mod=*/(short)0, c, /*reuse_a=*/false, /*reuse_b=*/false);
}

// Load a 16x32 bf16 fragment (A layout, also used for B^T row-major tiles).
// tile points at element [row 0, k 0] of the 16x32 tile; ld = row stride.
// Lane L holds row L%16; VGPR0..3 hold k = 8g..8g+7, VGPR4..7 hold k = 16+8g..
__device__ __forceinline__ v16bf ld_frag(const __bf16* tile, long ld, int lane) {
  const int r = lane & 15, g = lane >> 4;
  const __bf16* p = tile + (long)r * ld + g * 8;
  v16bf f;
  __builtin_memcpy(&f, p, 16);                                    // k = 8g..8g+7
  __builtin_memcpy(reinterpret_cast<char*>(&f) + 16, p + 16, 16); // k = 16+8g..
  return f;
}

// ---------------------------------------------------------------------------
// Kernel 1/2: rms-norm rows of f32 [rows, D] -> bf16.  One row per block.
// ---------------------------------------------------------------------------
__global__ void k_rmsnorm_bf16(const float* __restrict__ x,
                               __bf16* __restrict__ y) {
  const long row = blockIdx.x;
  const float* xr = x + row * DD;
  __bf16* yr = y + row * DD;

  float ss = 0.0f;
  for (int i = threadIdx.x * 4; i < DD; i += blockDim.x * 4) {
    float4 v = *(const float4*)(xr + i);
    ss += v.x * v.x + v.y * v.y + v.z * v.z + v.w * v.w;
  }
  for (int off = 16; off > 0; off >>= 1) ss += __shfl_xor(ss, off, 32);
  __shared__ float red[8];
  if ((threadIdx.x & 31) == 0) red[threadIdx.x >> 5] = ss;
  __syncthreads();
  if (threadIdx.x == 0) {
    float t = 0.0f;
    for (int j = 0; j < 8; ++j) t += red[j];
    red[0] = t;
  }
  __syncthreads();
  const float rstd = rsqrtf(red[0] * (1.0f / (float)DD) + 1e-6f);

  for (int i = threadIdx.x * 4; i < DD; i += blockDim.x * 4) {
    float4 v = *(const float4*)(xr + i);
    v4bf o;
    o.x = f2bf(v.x * rstd); o.y = f2bf(v.y * rstd);
    o.z = f2bf(v.z * rstd); o.w = f2bf(v.w * rstd);
    *(v4bf*)(yr + i) = o;
  }
}

// ---------------------------------------------------------------------------
// Kernel 3: transpose task_n [B,T,D] bf16 -> TnT [B, D, 64] (t padded to 64).
// ---------------------------------------------------------------------------
__global__ void k_transpose_tasks(const __bf16* __restrict__ Tn,
                                  __bf16* __restrict__ TnT) {
  const int b = blockIdx.x;
  const __bf16* src = Tn + (long)b * TT * DD;
  __bf16* dst = TnT + (long)b * DD * 64;
  const __bf16 zero = f2bf(0.0f);
  for (int idx = threadIdx.x; idx < DD * 64; idx += blockDim.x) {
    const int d = idx >> 6, t = idx & 63;
    dst[idx] = (t < TT) ? src[(long)t * DD + d] : zero;
  }
}

// ---------------------------------------------------------------------------
// Kernel 4: fused attention.  Each wave owns 16 query rows:
//   logits[16,48] = Pn_tile @ Tn^T   (3 WMMA col-tiles, K=2048)
//   softmax over T with mask bias    (shfl reductions, width 16)
//   attn staged bf16 in LDS [16,64]  (t padded with zeros)
//   q_raw[16,2048] = attn @ task_n   (B-fragments from TnT, K=64)
// ---------------------------------------------------------------------------
__global__ void k_attn_queries(const __bf16* __restrict__ Pn,
                               const __bf16* __restrict__ Tn,
                               const __bf16* __restrict__ TnT,
                               const unsigned char* __restrict__ mask,
                               __bf16* __restrict__ Q) {
  __shared__ __bf16 attn_smem[8][16 * 64];   // 16 KB

  const int lane = threadIdx.x & 31;
  const int w    = threadIdx.x >> 5;
  const int r    = lane & 15, g = lane >> 4;

  const int n_global = (blockIdx.x * 8 + w) * 16;    // 128 | 768 -> no straddle
  const int b  = n_global / NN;
  const int nb = n_global % NN;

  const __bf16* Ap = Pn  + ((long)b * NN + nb) * DD;
  const __bf16* Tp = Tn  + (long)b * TT * DD;
  const __bf16* TT_b = TnT + (long)b * DD * 64;
  __bf16* Qout = Q + ((long)b * NN + nb) * DD;

  // ---- logits: [16 x 48] in three f32 accumulators -------------------------
  v8f acc0 = {}, acc1 = {}, acc2 = {};
  for (int k = 0; k < DD; k += 32) {
    v16bf a  = ld_frag(Ap + k, DD, lane);
    v16bf b0 = ld_frag(Tp + (long)0  * DD + k, DD, lane);
    v16bf b1 = ld_frag(Tp + (long)16 * DD + k, DD, lane);
    v16bf b2 = ld_frag(Tp + (long)32 * DD + k, DD, lane);
    acc0 = wmma_bf16(a, b0, acc0);
    acc1 = wmma_bf16(a, b1, acc1);
    acc2 = wmma_bf16(a, b2, acc2);
  }

  // ---- masked, scaled softmax over T --------------------------------------
  const float scale = 0.022097086912079608f;   // 1/sqrt(2048)
  const float negbig = -3.4028234e38f;
  const float bias0 = mask[b * TT + 0  + r] ? 0.0f : negbig;
  const float bias1 = mask[b * TT + 16 + r] ? 0.0f : negbig;
  const float bias2 = mask[b * TT + 32 + r] ? 0.0f : negbig;

  #pragma unroll
  for (int i = 0; i < 8; ++i) {
    float x0 = acc0[i] * scale + bias0;
    float x1 = acc1[i] * scale + bias1;
    float x2 = acc2[i] * scale + bias2;
    float mx = fmaxf(fmaxf(x0, x1), x2);
    for (int off = 8; off > 0; off >>= 1) mx = fmaxf(mx, __shfl_xor(mx, off, 16));
    float e0 = __expf(x0 - mx), e1 = __expf(x1 - mx), e2 = __expf(x2 - mx);
    float s = e0 + e1 + e2;
    for (int off = 8; off > 0; off >>= 1) s += __shfl_xor(s, off, 16);
    const float inv = 1.0f / (s + 1e-30f);
    const int row = i + 8 * g;
    attn_smem[w][row * 64 + 0  + r] = f2bf(e0 * inv);
    attn_smem[w][row * 64 + 16 + r] = f2bf(e1 * inv);
    attn_smem[w][row * 64 + 32 + r] = f2bf(e2 * inv);
    attn_smem[w][row * 64 + 48 + r] = f2bf(0.0f);   // K padding
  }
  __syncthreads();

  // ---- q_raw = attn @ task_n (K = 64, padded) ------------------------------
  const v16bf a0 = ld_frag(&attn_smem[w][0],  64, lane);   // k = t 0..31
  const v16bf a1 = ld_frag(&attn_smem[w][32], 64, lane);   // k = t 32..63

  for (int d0 = 0; d0 < DD; d0 += 16) {
    v16bf b0 = ld_frag(TT_b + (long)d0 * 64,      64, lane);
    v16bf b1 = ld_frag(TT_b + (long)d0 * 64 + 32, 64, lane);
    v8f acc = {};
    acc = wmma_bf16(a0, b0, acc);
    acc = wmma_bf16(a1, b1, acc);
    const int col = d0 + r;
    #pragma unroll
    for (int i = 0; i < 8; ++i)
      Qout[(long)(i + 8 * g) * DD + col] = f2bf(acc[i]);
  }
}

// ---------------------------------------------------------------------------
// Kernel 5: score sweep + argmax + one-hot write.
// Each wave owns 32 query rows (2 row-tiles) and sweeps m in steps of 64
// (4 col-tiles): 8 accumulators, 8 WMMAs per 6 fragment loads per k-step.
// Argmax ties resolve to the lowest index (JAX semantics).
// ---------------------------------------------------------------------------
__global__ void k_score_argmax(const __bf16* __restrict__ Q,
                               const __bf16* __restrict__ Pn,
                               float* __restrict__ out) {
  const int lane = threadIdx.x & 31;
  const int w    = threadIdx.x >> 5;
  const int r    = lane & 15, g = lane >> 4;

  const int n_global = (blockIdx.x * 8 + w) * 32;    // 256 | 768 -> no straddle
  const int b  = n_global / NN;
  const int nb = n_global % NN;

  const __bf16* Qb = Q  + ((long)b * NN + nb) * DD;  // 32 query rows
  const __bf16* Pb = Pn + (long)b * NN * DD;

  float bestv[2][8];
  int   bestm[2][8];
  #pragma unroll
  for (int rt = 0; rt < 2; ++rt)
    #pragma unroll
    for (int i = 0; i < 8; ++i) { bestv[rt][i] = -3.4028235e38f; bestm[rt][i] = 0; }

  for (int m0 = 0; m0 < NN; m0 += 64) {
    if (m0 + 64 < NN)   // hint the next B panel toward L2/WGP$
      __builtin_prefetch(Pb + (long)(m0 + 64 + r) * DD, 0, 3);

    v8f s00 = {}, s01 = {}, s02 = {}, s03 = {};
    v8f s10 = {}, s11 = {}, s12 = {}, s13 = {};
    #pragma unroll 2
    for (int k = 0; k < DD; k += 32) {
      v16bf a0 = ld_frag(Qb + k,                 DD, lane);
      v16bf a1 = ld_frag(Qb + (long)16 * DD + k, DD, lane);
      v16bf p0 = ld_frag(Pb + (long)(m0 +  0) * DD + k, DD, lane);
      v16bf p1 = ld_frag(Pb + (long)(m0 + 16) * DD + k, DD, lane);
      v16bf p2 = ld_frag(Pb + (long)(m0 + 32) * DD + k, DD, lane);
      v16bf p3 = ld_frag(Pb + (long)(m0 + 48) * DD + k, DD, lane);
      s00 = wmma_bf16(a0, p0, s00);
      s01 = wmma_bf16(a0, p1, s01);
      s02 = wmma_bf16(a0, p2, s02);
      s03 = wmma_bf16(a0, p3, s03);
      s10 = wmma_bf16(a1, p0, s10);
      s11 = wmma_bf16(a1, p1, s11);
      s12 = wmma_bf16(a1, p2, s12);
      s13 = wmma_bf16(a1, p3, s13);
    }
    // candidates scanned in increasing m with strict '>' => lowest idx on tie
    #pragma unroll
    for (int i = 0; i < 8; ++i) {
      {
        float v = s00[i]; int c = m0 + r;
        if (s01[i] > v) { v = s01[i]; c = m0 + 16 + r; }
        if (s02[i] > v) { v = s02[i]; c = m0 + 32 + r; }
        if (s03[i] > v) { v = s03[i]; c = m0 + 48 + r; }
        if (v > bestv[0][i]) { bestv[0][i] = v; bestm[0][i] = c; }
      }
      {
        float v = s10[i]; int c = m0 + r;
        if (s11[i] > v) { v = s11[i]; c = m0 + 16 + r; }
        if (s12[i] > v) { v = s12[i]; c = m0 + 32 + r; }
        if (s13[i] > v) { v = s13[i]; c = m0 + 48 + r; }
        if (v > bestv[1][i]) { bestv[1][i] = v; bestm[1][i] = c; }
      }
    }
  }

  // cross-lane argmax within each 16-lane row group, then write one-hot rows
  #pragma unroll
  for (int rt = 0; rt < 2; ++rt) {
    #pragma unroll
    for (int i = 0; i < 8; ++i) {
      float v = bestv[rt][i]; int m = bestm[rt][i];
      for (int off = 8; off > 0; off >>= 1) {
        float ov = __shfl_xor(v, off, 16);
        int   om = __shfl_xor(m, off, 16);
        if (ov > v || (ov == v && om < m)) { v = ov; m = om; }
      }
      const int n_row = nb + 16 * rt + i + 8 * g;
      float* orow = out + ((long)b * NN + n_row) * (long)NN;
      for (int j = r; j < NN; j += 16)
        orow[j] = (j == m) ? 1.0f : 0.0f;
    }
  }
}

// ---------------------------------------------------------------------------
// Host launcher
// ---------------------------------------------------------------------------
extern "C" void kernel_launch(void* const* d_in, const int* in_sizes, int n_in,
                              void* d_out, int out_size, void* d_ws, size_t ws_size,
                              hipStream_t stream) {
  const float* patches = (const float*)d_in[0];              // [B,N,D] f32
  const float* tasks   = (const float*)d_in[1];              // [B,T,D] f32
  const unsigned char* mask = (const unsigned char*)d_in[2]; // [B,T] bool

  // bf16 workspace layout (~108 MB)
  __bf16* Pn  = (__bf16*)d_ws;                            // [B,N,D]
  __bf16* Tn  = Pn  + (long)BB * NN * DD;                 // [B,T,D]
  __bf16* TnT = Tn  + (long)BB * TT * DD;                 // [B,D,64]
  __bf16* Qw  = TnT + (long)BB * DD * 64;                 // [B,N,D]

  k_rmsnorm_bf16<<<BB * NN, 256, 0, stream>>>(patches, Pn);
  k_rmsnorm_bf16<<<BB * TT, 256, 0, stream>>>(tasks, Tn);
  k_transpose_tasks<<<BB, 256, 0, stream>>>(Tn, TnT);

  // attention: 8 waves/block, 16 rows/wave -> 128 rows/block
  k_attn_queries<<<(BB * NN) / 128, 256, 0, stream>>>(Pn, Tn, TnT, mask, Qw);
  // score: 8 waves/block, 32 rows/wave -> 256 rows/block
  k_score_argmax<<<(BB * NN) / 256, 256, 0, stream>>>(Qw, Pn, (float*)d_out);
}
